// LSTMModel_71313636982856
// MI455X (gfx1250) — compile-verified, matrix operand order checked by reading, not run on hardware
//
#include <hip/hip_runtime.h>

typedef __bf16 bf16;
typedef __attribute__((ext_vector_type(16))) __bf16 v16bf;
typedef __attribute__((ext_vector_type(8)))  float  v8f;
typedef __attribute__((ext_vector_type(4)))  float  v4f;

#define T_LEN 2048
#define HID   128
#define STEP_BYTES 4096          // 4 A-fragments * 1024B per timestep in workspace

struct Params {
  const float* x;
  const float* Wih[3];
  const float* Whh[3];
  const float* bih[3];
  const float* bhh[3];
  const float* Wc;
  const float* bc;
  float* out;
  char*  ws;
};

// Fast gate math: v_exp_f32 + v_rcp_f32 (TRANS ops co-execute with WMMA).
__device__ __forceinline__ float sigm(float x) {
  return __builtin_amdgcn_rcpf(1.f + __expf(-x));
}
__device__ __forceinline__ float tanh_f(float x) {
  return 1.f - 2.f * __builtin_amdgcn_rcpf(1.f + __expf(2.f * x));
}

// ISA 16-bit A/B fragment K map: lane<16 -> K = e<8 ? e : e+8 ; lane>=16 -> +8/+16.
// Load one lane-slot (16 bf16) of a 16x32 weight fragment from row-major f32 W[n][k].
__device__ __forceinline__ v16bf load_w_frag(const float* W, int n, int kbase, int ls) {
  const float* p = W + (size_t)n * HID + kbase + ((ls < 16) ? 0 : 8);
  v4f a0 = *(const v4f*)(p);
  v4f a1 = *(const v4f*)(p + 4);
  v4f a2 = *(const v4f*)(p + 16);
  v4f a3 = *(const v4f*)(p + 20);
  v16bf r;
#pragma unroll
  for (int i = 0; i < 4; ++i) {
    r[i]      = (bf16)a0[i];
    r[4 + i]  = (bf16)a1[i];
    r[8 + i]  = (bf16)a2[i];
    r[12 + i] = (bf16)a3[i];
  }
  return r;
}

// MODE 0: layer 0 (VALU input projection K=4, writes h-seq)
// MODE 1: middle layer (WMMA input GEMM from ws, writes h-seq in place)
// MODE 2: last layer  (WMMA input GEMM from ws, no h-seq write)
template <int MODE>
__device__ __forceinline__ void do_layer(
    const float* __restrict__ Wih, const float* __restrict__ Whh,
    const float* __restrict__ bih, const float* __restrict__ bhh,
    const float* __restrict__ x, char* __restrict__ wsBase,
    v16bf* hbufFrag, v4f* xbuf,
    int tid, int lane, int w, int c, int hiL, int bbase) {
  // ---- per-layer register-resident state ----
  v16bf whhF[16];          // recurrent weights, 16 fragments
  v16bf wihF[16];          // input weights (MODE>0), 16 fragments
  float biasv[4];
  v4f   w0[4];             // MODE==0 input weights (K=4)
#pragma unroll
  for (int g = 0; g < 4; ++g) {
    int n = g * HID + w * 16 + c;     // gate-g row for this lane's output column
    biasv[g] = bih[n] + bhh[n];
#pragma unroll
    for (int kt = 0; kt < 4; ++kt) {
      whhF[g * 4 + kt] = load_w_frag(Whh, n, kt * 32, lane);
      if constexpr (MODE > 0) wihF[g * 4 + kt] = load_w_frag(Wih, n, kt * 32, lane);
    }
    if constexpr (MODE == 0) w0[g] = *(const v4f*)(Wih + (size_t)n * 4);
  }
  __syncthreads();   // previous layer's LDS reads complete before re-init
  {
    uint4 z; z.x = 0u; z.y = 0u; z.z = 0u; z.w = 0u;
    ((uint4*)hbufFrag)[tid] = z;     // h0 = 0 (A-fragment layout)
  }
  v8f cF;
#pragma unroll
  for (int v = 0; v < 8; ++v) cF[v] = 0.f;
  __syncthreads();

  // h scatter coordinates: this wave owns hidden units j = w*16 + c.
  const int ktw = w >> 1;
  const int e0  = (c & 7) + 8 * (w & 1);
  const int ld0 = (c >= 8) ? 16 : 0;
  bf16* hb = (bf16*)hbufFrag;

  for (int t = 0; t < T_LEN; ++t) {
    if constexpr (MODE == 0) {
      if (tid < 16)
        xbuf[tid] = *(const v4f*)(x + ((size_t)(bbase + tid) * T_LEN + t) * 4);
    }
    v16bf aH[4];
#pragma unroll
    for (int kt = 0; kt < 4; ++kt) aH[kt] = hbufFrag[kt * 32 + lane];
    v16bf aX[4];
    if constexpr (MODE > 0) {
      const char* xp = wsBase + (size_t)t * STEP_BYTES;
#pragma unroll
      for (int kt = 0; kt < 4; ++kt)
        aX[kt] = *(const v16bf*)(xp + (size_t)kt * 1024 + lane * 32);
      __builtin_prefetch(wsBase + (size_t)(t + 1) * STEP_BYTES + tid * 16, 0, 1);
    }
    __syncthreads();   // all h-buffer reads complete before rewrite

    // accumulators: bias (+ layer-0 input projection)
    v8f acc[4];
#pragma unroll
    for (int g = 0; g < 4; ++g) {
#pragma unroll
      for (int v = 0; v < 8; ++v) acc[g][v] = biasv[g];
    }
    if constexpr (MODE == 0) {
#pragma unroll
      for (int v = 0; v < 8; ++v) {
        int m = v + 8 * hiL;
        v4f xm = xbuf[m];
#pragma unroll
        for (int g = 0; g < 4; ++g)
          acc[g][v] += xm[0] * w0[g][0] + xm[1] * w0[g][1] +
                       xm[2] * w0[g][2] + xm[3] * w0[g][3];
      }
    }
    // Recurrent GEMM: 16 register-operand WMMAs, 4 independent chains per round.
#pragma unroll
    for (int kt = 0; kt < 4; ++kt) {
#pragma unroll
      for (int g = 0; g < 4; ++g)
        acc[g] = __builtin_amdgcn_wmma_f32_16x16x32_bf16(
            false, aH[kt], false, whhF[g * 4 + kt], (short)0, acc[g], false, false);
    }
    // Input GEMM (MODE>0): 16 more register-operand WMMAs.
    if constexpr (MODE > 0) {
#pragma unroll
      for (int kt = 0; kt < 4; ++kt) {
#pragma unroll
        for (int g = 0; g < 4; ++g)
          acc[g] = __builtin_amdgcn_wmma_f32_16x16x32_bf16(
              false, aX[kt], false, wihF[g * 4 + kt], (short)0, acc[g], false, false);
      }
    }
    // Gates + cell update; scatter new h into A-fragment layout.
#pragma unroll
    for (int v = 0; v < 8; ++v) {
      float iv = sigm(acc[0][v]);
      float fv = sigm(acc[1][v]);
      float gv = tanh_f(acc[2][v]);
      float ov = sigm(acc[3][v]);
      float cv = fv * cF[v] + iv * gv;
      cF[v] = cv;
      float hv = ov * tanh_f(cv);
      int m = v + 8 * hiL;
      hb[ktw * 512 + (m + ld0) * 16 + e0] = (bf16)hv;
    }
    __syncthreads();   // new h visible to all waves
    if constexpr (MODE < 2) {
      uint4 hq = ((const uint4*)hbufFrag)[tid];
      *(uint4*)(wsBase + (size_t)t * STEP_BYTES + tid * 16) = hq;
    }
  }
}

__global__ __launch_bounds__(256, 1)
void lstm3_wmma_kernel(Params p) {
  __shared__ v16bf hbufFrag[128];   // 4 fragments * 32 slots = 4 KB
  __shared__ v4f   xbuf[16];        // layer-0 x staging

  const int tid  = threadIdx.x;
  const int lane = tid & 31;        // wave32
  const int w    = tid >> 5;        // wave id 0..7
  const int c    = lane & 15;
  const int hiL  = lane >> 4;
  const int bbase = blockIdx.x * 16;
  char* wsBase = p.ws + (size_t)blockIdx.x * ((size_t)T_LEN * STEP_BYTES);

  do_layer<0>(p.Wih[0], p.Whh[0], p.bih[0], p.bhh[0], p.x, wsBase,
              hbufFrag, xbuf, tid, lane, w, c, hiL, bbase);
  do_layer<1>(p.Wih[1], p.Whh[1], p.bih[1], p.bhh[1], p.x, wsBase,
              hbufFrag, xbuf, tid, lane, w, c, hiL, bbase);
  do_layer<2>(p.Wih[2], p.Whh[2], p.bih[2], p.bhh[2], p.x, wsBase,
              hbufFrag, xbuf, tid, lane, w, c, hiL, bbase);

  // Final classifier: logits[b] = h_last[b,:] . Wc + bc   (h in A-fragment layout)
  if (tid < 16) {
    const bf16* hb = (const bf16*)hbufFrag;
    float acc = 0.f;
    for (int k = 0; k < HID; ++k) {
      int kt = k >> 5, kl = k & 31;
      int ld, e;
      if (kl < 8)       { ld = tid;      e = kl;      }
      else if (kl < 16) { ld = tid + 16; e = kl - 8;  }
      else if (kl < 24) { ld = tid;      e = kl - 8;  }
      else              { ld = tid + 16; e = kl - 16; }
      acc += (float)hb[kt * 512 + ld * 16 + e] * p.Wc[k];
    }
    p.out[bbase + tid] = acc + p.bc[0];
  }
}

extern "C" void kernel_launch(void* const* d_in, const int* in_sizes, int n_in,
                              void* d_out, int out_size, void* d_ws, size_t ws_size,
                              hipStream_t stream) {
  (void)in_sizes; (void)n_in; (void)out_size; (void)ws_size;
  Params p;
  p.x      = (const float*)d_in[0];
  p.Wih[0] = (const float*)d_in[1];
  p.Whh[0] = (const float*)d_in[2];
  p.bih[0] = (const float*)d_in[3];
  p.bhh[0] = (const float*)d_in[4];
  p.Wih[1] = (const float*)d_in[5];
  p.Whh[1] = (const float*)d_in[6];
  p.bih[1] = (const float*)d_in[7];
  p.bhh[1] = (const float*)d_in[8];
  p.Wih[2] = (const float*)d_in[9];
  p.Whh[2] = (const float*)d_in[10];
  p.bih[2] = (const float*)d_in[11];
  p.bhh[2] = (const float*)d_in[12];
  p.Wc     = (const float*)d_in[13];
  p.bc     = (const float*)d_in[14];
  p.out    = (float*)d_out;
  p.ws     = (char*)d_ws;   // 8 WGs * 2048 steps * 4KB = 64 MB scratch

  lstm3_wmma_kernel<<<dim3(8), dim3(256), 0, stream>>>(p);
}
